// SparseSelfAttention_75625784148411
// MI455X (gfx1250) — matrix operand order repaired and checked by reading, not compiled, not run on hardware
//
#include <hip/hip_runtime.h>

// ---------------------------------------------------------------------------
// Problem constants (from reference): B=16, N=2048, D=256, H=64, EPS=1e-5
// ---------------------------------------------------------------------------
#define B_   16
#define N_   2048
#define D_   256
#define H_   64
#define NBLK 30          // output rows per attention workgroup (32-row slab w/ 1-row halo)
#define SCOL 2048        // columns of the S slab

typedef __attribute__((ext_vector_type(16))) _Float16 v16h;
typedef __attribute__((ext_vector_type(8)))  _Float16 v8h;
typedef __attribute__((ext_vector_type(8)))  float    v8f;

__device__ inline v8f wmma_f16(v16h a, v16h b, v8f c) {
  // D = A(16x32 f16) * B(32x16 f16) + C(16x16 f32)
  return __builtin_amdgcn_wmma_f32_16x16x32_f16(
      /*neg_a=*/false, a, /*neg_b=*/false, b,
      /*c_mod=*/(short)0, c, /*reuse_a=*/false, /*reuse_b=*/false);
}

// ---------------------------------------------------------------------------
// Kernel 0: one-time prep — transpose + convert Wq/Wk/Wv (D x H fp32) into
// Wt[proj][h][d] f16 so B-fragment lanes read 16 *contiguous* K values.
// ---------------------------------------------------------------------------
__global__ __launch_bounds__(256)
void prep_w(const float* __restrict__ Wq, const float* __restrict__ Wk,
            const float* __restrict__ Wv, _Float16* __restrict__ Wt) {
  const int proj = blockIdx.x;
  const float* W = (proj == 0) ? Wq : (proj == 1) ? Wk : Wv;
  _Float16* o = Wt + (size_t)proj * H_ * D_;
  for (int idx = threadIdx.x; idx < H_ * D_; idx += 256) {
    const int h = idx >> 8;          // D_ == 256
    const int d = idx & (D_ - 1);
    o[idx] = (_Float16)W[(size_t)d * H_ + h];
  }
}

// ---------------------------------------------------------------------------
// Kernel 1: fused QKV projection.  One wave computes a 16x64 tile of one
// projection for one batch.  The x tile (16x256 fp32, 16 KB) is staged into
// LDS with gfx1250 async-to-LDS 128-bit copies (ASYNCcnt), A/B fragments are
// contiguous 128-bit loads, all math via v_wmma_f32_16x16x32_f16.
// q/k stored [n][h] f16 (A / B friendly); v stored transposed [h][m] f16.
// ---------------------------------------------------------------------------
__global__ __launch_bounds__(32)
void qkv_kernel(const float* __restrict__ x,
                const float* __restrict__ bq, const float* __restrict__ bk,
                const float* __restrict__ bv,
                const _Float16* __restrict__ Wt,
                _Float16* __restrict__ qh, _Float16* __restrict__ kh,
                _Float16* __restrict__ vT) {
  __shared__ float xs[16 * D_];        // 16 KB x tile

  const int tile = blockIdx.x;         // N/16 row tiles
  const int b    = blockIdx.y;         // batch
  const int proj = blockIdx.z;         // 0=q 1=k 2=v

  const _Float16* W  = Wt + (size_t)proj * H_ * D_;
  const float*    bi = (proj == 0) ? bq : (proj == 1) ? bk : bv;

  const int n0   = tile * 16;
  const int lane = threadIdx.x;
  const int lg   = lane >> 4;          // lane group (0: lanes 0-15, 1: 16-31)
  const int lr   = lane & 15;

  // ---- async-stage x[b][n0..n0+15][0..255] into LDS (1024 x b128) ----------
  const float* xt = x + ((size_t)b * N_ + n0) * D_;
  for (int i = 0; i < 32; ++i) {
    const int e = i * 32 + lane;                       // float4 index
    const unsigned lds = (unsigned)(size_t)(xs + e * 4);
    const unsigned long long ga = (unsigned long long)(size_t)(xt + e * 4);
    asm volatile("global_load_async_to_lds_b128 %0, %1, off"
                 :: "v"(lds), "v"(ga) : "memory");
  }
  asm volatile("s_wait_asynccnt 0" ::: "memory");      // single wave: no barrier

  v8f c[4] = {{}, {}, {}, {}};

  const float* xrow = xs + lr * D_;                    // this lane's A row
  for (int ks = 0; ks < D_ / 32; ++ks) {
    const int kbase = ks * 32;
    // A fragment (contiguous layout): halves 0..7 -> K = kbase+16lg..+7,
    //                                 halves 8..15 -> K = kbase+16lg+8..+15
    const float4* p0 = (const float4*)(xrow + kbase + 16 * lg);
    const float4* p1 = (const float4*)(xrow + kbase + 16 * lg + 8);
    float4 f0 = p0[0], f1 = p0[1], f2 = p1[0], f3 = p1[1];
    v16h a;
    a[0]  = (_Float16)f0.x; a[1]  = (_Float16)f0.y; a[2]  = (_Float16)f0.z; a[3]  = (_Float16)f0.w;
    a[4]  = (_Float16)f1.x; a[5]  = (_Float16)f1.y; a[6]  = (_Float16)f1.z; a[7]  = (_Float16)f1.w;
    a[8]  = (_Float16)f2.x; a[9]  = (_Float16)f2.y; a[10] = (_Float16)f2.z; a[11] = (_Float16)f2.w;
    a[12] = (_Float16)f3.x; a[13] = (_Float16)f3.y; a[14] = (_Float16)f3.z; a[15] = (_Float16)f3.w;

    #pragma unroll
    for (int nt = 0; nt < 4; ++nt) {
      // B fragment: B[kk][n] = W[kbase+kk][nt*16+n] = Wt[nt*16+n][kbase+kk]
      // lane lr = column n, lg = K half; 16 contiguous f16 -> two b128 loads.
      const _Float16* wp = W + (size_t)(nt * 16 + lr) * D_ + kbase + 16 * lg;
      v8h lo = ((const v8h*)wp)[0];
      v8h hi = ((const v8h*)wp)[1];
      v16h bf;
      #pragma unroll
      for (int h = 0; h < 8; ++h) { bf[h] = lo[h]; bf[8 + h] = hi[h]; }
      c[nt] = wmma_f16(a, bf, c[nt]);
    }
  }

  // Bias add + f16 store. C layout: VGPR r -> row M = r + 8*lg, col N = lr.
  if (proj == 2) {
    // store v transposed: vT[b][h][m]; lane's 8 rows are contiguous in m.
    #pragma unroll
    for (int nt = 0; nt < 4; ++nt) {
      const float bvv = bi[nt * 16 + lr];
      v8h sv;
      #pragma unroll
      for (int r = 0; r < 8; ++r) sv[r] = (_Float16)(c[nt][r] + bvv);
      *(v8h*)(vT + ((size_t)b * H_ + nt * 16 + lr) * N_ + n0 + 8 * lg) = sv;
    }
  } else {
    _Float16* out = (proj == 0) ? qh : kh;
    #pragma unroll
    for (int nt = 0; nt < 4; ++nt) {
      const float bvv = bi[nt * 16 + lr];
      _Float16* op = out + ((size_t)b * N_ + n0 + 8 * lg) * H_ + nt * 16 + lr;
      #pragma unroll
      for (int r = 0; r < 8; ++r) op[(size_t)r * H_] = (_Float16)(c[nt][r] + bvv);
    }
  }
}

// ---------------------------------------------------------------------------
// Kernel 2: fused attention block.  256 threads (8 waves).  Each workgroup
// owns 30 output rows of one batch:
//   Phase 1: S[32][2048] = scale * q_ext @ k^T   (WMMA, slab kept in LDS)
//   Phase 2: in-place  S <- expm1(relu(S - sigmoid(conv3x3(S)+cb)))
//   Phase 3: row sums -> 1/(sum+eps)
//   Phase 4: O = P_norm @ v (WMMA, K split across waves, ds_add_f32 reduce)
//   Phase 5: write 30x64 fp32 rows
// ---------------------------------------------------------------------------
__global__ __launch_bounds__(256)
void attn_kernel(const _Float16* __restrict__ qh,
                 const _Float16* __restrict__ kh,
                 const _Float16* __restrict__ vT,
                 const float* __restrict__ conv_w,
                 const float* __restrict__ conv_b,
                 float* __restrict__ out) {
  extern __shared__ float smem[];
  float* S       = smem;                    // 32 * 2048 floats (256 KB)
  float* O       = smem + 32 * SCOL;        // 32 * 64 floats
  float* rowinv  = O + 32 * 64;             // 32 floats
  float* savecol = rowinv + 32;             // 32 floats

  const int tid  = threadIdx.x;
  const int wave = tid >> 5;
  const int lane = tid & 31;
  const int lg   = lane >> 4;
  const int lr   = lane & 15;
  const int b    = blockIdx.y;
  const int rowbase = blockIdx.x * NBLK - 1;     // slab row 0 = matrix row rowbase

  // zero the O accumulation buffer (visible after the phase-1 barrier)
  for (int e = tid; e < 32 * 64; e += 256) O[e] = 0.f;

  // -------------------- Phase 1: S = scale * q_ext @ k^T --------------------
  v16h aQ[2][2];
  #pragma unroll
  for (int t = 0; t < 2; ++t) {
    const int g  = rowbase + 16 * t + lr;
    const bool ok = (g >= 0) && (g < N_);
    const _Float16* qrow = qh + ((size_t)b * N_ + (ok ? g : 0)) * H_;
    #pragma unroll
    for (int s = 0; s < 2; ++s) {
      v16h a;
      if (ok) {
        v8h lo = *(const v8h*)(qrow + 32 * s + 16 * lg);
        v8h hi = *(const v8h*)(qrow + 32 * s + 16 * lg + 8);
        #pragma unroll
        for (int h = 0; h < 8; ++h) { a[h] = lo[h]; a[8 + h] = hi[h]; }
      } else {
        #pragma unroll
        for (int h = 0; h < 16; ++h) a[h] = (_Float16)0.f;
      }
      aQ[t][s] = a;
    }
  }

  const _Float16* kb = kh + (size_t)b * N_ * H_;
  const float scale = 0.125f;                 // H^{-1/2}
  for (int nt = 0; nt < 16; ++nt) {           // each wave covers 256 columns
    const int colbase = wave * 256 + nt * 16;
    if (nt + 1 < 16)                          // stream k rows through L2
      __builtin_prefetch(kb + (size_t)(colbase + 16 + lr) * H_, 0, 0);
    v8f c0 = {}, c1 = {};
    #pragma unroll
    for (int s = 0; s < 2; ++s) {
      // B fragment: B[kk][n] = k[colbase+n][kk]; lane lr = col, lg = K half.
      const _Float16* krow = kb + (size_t)(colbase + lr) * H_ + 32 * s + 16 * lg;
      v8h lo = ((const v8h*)krow)[0];
      v8h hi = ((const v8h*)krow)[1];
      v16h bf;
      #pragma unroll
      for (int h = 0; h < 8; ++h) { bf[h] = lo[h]; bf[8 + h] = hi[h]; }
      c0 = wmma_f16(aQ[0][s], bf, c0);
      c1 = wmma_f16(aQ[1][s], bf, c1);
    }
    #pragma unroll
    for (int r = 0; r < 8; ++r) {
      S[(size_t)(r + 8 * lg) * SCOL + colbase + lr]      = c0[r] * scale;
      S[(size_t)(16 + r + 8 * lg) * SCOL + colbase + lr] = c1[r] * scale;
    }
  }
  __syncthreads();

  // ---- Phase 2: in-place conv3x3 -> sigmoid -> relu -> expm1 (rows 1..30) ----
  const float w00 = conv_w[0], w01 = conv_w[1], w02 = conv_w[2];
  const float w10 = conv_w[3], w11 = conv_w[4], w12 = conv_w[5];
  const float w20 = conv_w[6], w21 = conv_w[7], w22 = conv_w[8];
  const float cb  = conv_b[0];

  float p[NBLK];
  for (int j = 0; j < 8; ++j) {               // 8 column chunks of 256
    const int c = j * 256 + tid;
    const bool hasL = (c > 0);
    const bool hasR = (c < SCOL - 1);
    auto getL = [&](int i) -> float {
      if (!hasL) return 0.f;
      if (tid == 0) return savecol[i];        // original col j*256-1 (j>0)
      return S[(size_t)i * SCOL + c - 1];
    };
    float l0 = getL(0), m0 = S[c],        r0 = hasR ? S[c + 1] : 0.f;
    float l1 = getL(1), m1 = S[SCOL + c], r1 = hasR ? S[SCOL + c + 1] : 0.f;
    #pragma unroll
    for (int i = 1; i <= NBLK; ++i) {
      float l2 = getL(i + 1);
      float m2 = S[(size_t)(i + 1) * SCOL + c];
      float r2 = hasR ? S[(size_t)(i + 1) * SCOL + c + 1] : 0.f;
      float cv = w00 * l0 + w01 * m0 + w02 * r0 +
                 w10 * l1 + w11 * m1 + w12 * r1 +
                 w20 * l2 + w21 * m2 + w22 * r2 + cb;
      float sg = 1.f / (1.f + __expf(-cv));
      float t  = m1 - sg;
      t = (t > 0.f) ? t : 0.f;
      p[i - 1] = __expf(t) - 1.f;             // expm1(relu(...))
      l0 = l1; m0 = m1; r0 = r1;
      l1 = l2; m1 = m2; r1 = r2;
    }
    __syncthreads();
    if (tid < 32) savecol[tid] = S[(size_t)tid * SCOL + j * 256 + 255];
    __syncthreads();
    #pragma unroll
    for (int i = 1; i <= NBLK; ++i) S[(size_t)i * SCOL + c] = p[i - 1];
    __syncthreads();
  }

  // -------------------- Phase 3: zero-softmax denominators --------------------
  for (int i = 1 + wave; i <= NBLK; i += 8) {
    float s = 0.f;
    for (int k2 = lane; k2 < SCOL; k2 += 32) s += S[(size_t)i * SCOL + k2];
    #pragma unroll
    for (int off = 16; off > 0; off >>= 1) s += __shfl_xor(s, off, 32);
    if (lane == 0) rowinv[i] = 1.f / (s + 1e-5f);
  }
  if (tid == 0) { rowinv[0] = 0.f; rowinv[31] = 0.f; }
  __syncthreads();

  // -------------------- Phase 4: O = P_norm @ v (K split over waves) ----------
  v8f acc[2][4];
  #pragma unroll
  for (int t = 0; t < 2; ++t)
    #pragma unroll
    for (int nt = 0; nt < 4; ++nt) acc[t][nt] = (v8f){};

  const _Float16* vtb = vT + (size_t)b * H_ * N_;   // [h][m] layout
  for (int ksi = 0; ksi < 8; ++ksi) {
    const int mbase = wave * 256 + ksi * 32;
    v16h aP[2];
    #pragma unroll
    for (int t = 0; t < 2; ++t) {
      const int row = 16 * t + lr;
      const float ri = rowinv[row];
      v16h a;
      #pragma unroll
      for (int h = 0; h < 16; ++h) {
        const int k = mbase + h + 16 * lg;   // contiguous A-frag K map
        a[h] = (_Float16)(S[(size_t)row * SCOL + k] * ri);
      }
      aP[t] = a;
    }
    #pragma unroll
    for (int nt = 0; nt < 4; ++nt) {
      // B fragment from vT: B[kk][n] = vT[nt*16+n][mbase+kk]
      // lane lr = col n, lg = K half; 16 contiguous f16 -> two b128 loads.
      const _Float16* vp = vtb + (size_t)(nt * 16 + lr) * N_ + mbase + 16 * lg;
      v8h lo = ((const v8h*)vp)[0];
      v8h hi = ((const v8h*)vp)[1];
      v16h bf;
      #pragma unroll
      for (int h = 0; h < 8; ++h) { bf[h] = lo[h]; bf[8 + h] = hi[h]; }
      acc[0][nt] = wmma_f16(aP[0], bf, acc[0][nt]);
      acc[1][nt] = wmma_f16(aP[1], bf, acc[1][nt]);
    }
  }
  #pragma unroll
  for (int t = 0; t < 2; ++t)
    #pragma unroll
    for (int nt = 0; nt < 4; ++nt)
      #pragma unroll
      for (int r = 0; r < 8; ++r)
        atomicAdd(&O[(size_t)(16 * t + r + 8 * lg) * 64 + nt * 16 + lr],
                  acc[t][nt][r]);
  __syncthreads();

  // -------------------- Phase 5: write 30x64 fp32 output rows -----------------
  for (int e = tid; e < NBLK * 64; e += 256) {
    const int i = 1 + (e >> 6);
    const int h = e & 63;
    const int g = rowbase + i;
    if (g < N_) out[((size_t)b * N_ + g) * H_ + h] = O[(size_t)i * 64 + h];
  }
}

// ---------------------------------------------------------------------------
// Host launcher
// ---------------------------------------------------------------------------
extern "C" void kernel_launch(void* const* d_in, const int* in_sizes, int n_in,
                              void* d_out, int out_size, void* d_ws, size_t ws_size,
                              hipStream_t stream) {
  const float* x      = (const float*)d_in[0];
  const float* Wq     = (const float*)d_in[1];
  const float* bq     = (const float*)d_in[2];
  const float* Wk     = (const float*)d_in[3];
  const float* bk     = (const float*)d_in[4];
  const float* Wv     = (const float*)d_in[5];
  const float* bv     = (const float*)d_in[6];
  const float* conv_w = (const float*)d_in[7];
  const float* conv_b = (const float*)d_in[8];

  char* ws = (char*)d_ws;
  const size_t qkv_bytes = (size_t)B_ * N_ * H_ * sizeof(_Float16);  // 4 MB each
  _Float16* qh = (_Float16*)(ws);
  _Float16* kh = (_Float16*)(ws + qkv_bytes);
  _Float16* vT = (_Float16*)(ws + 2 * qkv_bytes);
  _Float16* Wt = (_Float16*)(ws + 3 * qkv_bytes);                    // 96 KB

  prep_w<<<dim3(3), 256, 0, stream>>>(Wq, Wk, Wv, Wt);

  dim3 g1(N_ / 16, B_, 3);
  qkv_kernel<<<g1, 32, 0, stream>>>(x, bq, bk, bv, Wt, qh, kh, vT);

  dim3 g2((N_ + NBLK - 1) / NBLK, B_);
  const size_t shmem = (size_t)(32 * SCOL + 32 * 64 + 64) * sizeof(float); // ~265 KB
  attn_kernel<<<g2, 256, shmem, stream>>>(qh, kh, vT, conv_w, conv_b, (float*)d_out);
}
